// GHMCLoss_42417097016426
// MI455X (gfx1250) — matrix operand order repaired and checked by reading, not compiled, not run on hardware
//
#include <hip/hip_runtime.h>

#define GHM_BINS 30
#define GHM_MOM  0.5f

typedef __attribute__((ext_vector_type(16))) _Float16     v16h;
typedef __attribute__((ext_vector_type(8)))  float        v8f;
typedef __attribute__((ext_vector_type(8)))  unsigned int v8u;
typedef __attribute__((ext_vector_type(4)))  float        v4f;

// Per-element math: one v_exp, one v_log, one v_rcp (all native TRANS ops).
__device__ __forceinline__ void ghm_elem(float x, float t, float& bce, int& idx) {
    float ax  = __builtin_fabsf(x);
    float e   = __expf(-ax);                      // v_mul + v_exp_f32
    float p   = 1.0f + e;                         // in (1,2]
    float r   = __builtin_amdgcn_rcpf(p);         // single v_rcp_f32 (~1 ulp)
    float sig = (x >= 0.0f) ? r : 1.0f - r;       // sigmoid(x)
    float g   = __builtin_fabsf(sig - t);
    int   b   = (int)(g * 30.0f);                 // floor, g >= 0
    idx = (b > GHM_BINS - 1) ? (GHM_BINS - 1) : b;
    // max(x,0) - x*t + log1p(exp(-|x|)); arg in (1,2] -> raw v_log_f32 is exact enough
    bce = fmaxf(x, 0.0f) - x * t + __builtin_amdgcn_logf(p);
}

__global__ void ghm_init(float* ws) {
    if (threadIdx.x < 64) ws[threadIdx.x] = 0.0f;   // gSum[32] ++ gCnt[32]
}

// Pass 1: single sweep over input/target. Per 32 elements per wave, two
// v_wmma_f32_16x16x32_f16 ops accumulate {per-bin BCE sum, per-bin count}
// into resident f32 accumulators (bins 0-15 and 16-31).
__global__ __launch_bounds__(256) void ghm_pass1(const float* __restrict__ in,
                                                 const float* __restrict__ tg,
                                                 float* __restrict__ gSum,
                                                 float* __restrict__ gCnt,
                                                 long long n) {
    __shared__ __align__(16) _Float16      sB[8][32];   // staged bce (f16) per wave
    __shared__ __align__(16) unsigned char sI[8][32];   // staged bin idx per wave
    __shared__ float sSum[32];
    __shared__ float sCnt[32];

    const int tid  = threadIdx.x;
    const int lane = tid & 31;
    const int w    = tid >> 5;
    const int n0   = lane & 15;          // output column / bin (low half)
    const int kb   = lane & 16;          // B-matrix K base: lanes0-15 K=0-15, lanes16-31 K=16-31

    if (tid < 32) { sSum[tid] = 0.0f; sCnt[tid] = 0.0f; }
    __syncthreads();

    v8f accLo = {};   // D[0,n]=bce sum bin n,   D[1,n]=count bin n      (bins 0-15)
    v8f accHi = {};   // same for bins 16-29 (cols 14,15 never hit -> stay 0)

    const long long stride = (long long)gridDim.x * blockDim.x * 4;
    const long long nMain  = n - (n % stride);
    long long i = ((long long)blockIdx.x * blockDim.x + tid) * 4;

    for (; i < nMain; i += stride) {
        v4f xv = __builtin_nontemporal_load((const v4f*)(in + i));
        v4f tv = __builtin_nontemporal_load((const v4f*)(tg + i));
        __builtin_prefetch(in + i + stride, 0, 1);   // global_prefetch_b8
        __builtin_prefetch(tg + i + stride, 0, 1);

        #pragma unroll
        for (int c = 0; c < 4; ++c) {
            float bce; int idx;
            ghm_elem(xv[c], tv[c], bce, idx);

            // Stage this wave's 32 elements to LDS (wave-lockstep; LDS ops
            // are in-order within a wave, no barrier needed).
            sB[w][lane] = (_Float16)bce;
            sI[w][lane] = (unsigned char)idx;
            __builtin_amdgcn_wave_barrier();

            // ---- Build B (32x16 one-hot): column n0, K = kb..kb+15 ----
            uint4 I = *(const uint4*)&sI[w][kb];
            unsigned int iw4[4] = {I.x, I.y, I.z, I.w};
            v8u bLo, bHi;
            #pragma unroll
            for (int v = 0; v < 8; ++v) {            // VGPR v holds K = kb+2v, kb+2v+1
                unsigned int word = iw4[v >> 1];
                unsigned int sh   = (v & 1) * 16;
                unsigned int b0   = (word >> sh) & 0xFFu;
                unsigned int b1   = (word >> (sh + 8)) & 0xFFu;
                bLo[v] = (b0 == (unsigned)n0        ? 0x3C00u : 0u) |
                         (b1 == (unsigned)n0        ? 0x3C000000u : 0u);
                bHi[v] = (b0 == (unsigned)(n0 + 16) ? 0x3C00u : 0u) |
                         (b1 == (unsigned)(n0 + 16) ? 0x3C000000u : 0u);
            }

            // ---- Build A (16x32): row0 = bce, row1 = 1.0, rows 2-15 = 0 ----
            // Lane m holds row m: lanes0-15 K=0-15(V0-7);
            //                     lanes16-31 K=16-31.
            v8u au;
            if (n0 == 0) {
                int o = (lane >> 4) * 16;            // 0 (lane0) or 16 (lane16)
                uint4 lo = *(const uint4*)&sB[w][o];
                uint4 hi = *(const uint4*)&sB[w][o + 8];
                au[0] = lo.x; au[1] = lo.y; au[2] = lo.z; au[3] = lo.w;
                au[4] = hi.x; au[5] = hi.y; au[6] = hi.z; au[7] = hi.w;
            } else if (n0 == 1) {
                #pragma unroll
                for (int v = 0; v < 8; ++v) au[v] = 0x3C003C00u;  // packed {1,1}
            } else {
                #pragma unroll
                for (int v = 0; v < 8; ++v) au[v] = 0u;
            }
            v16h A = __builtin_bit_cast(v16h, au);

            // EXEC is all-ones here (uniform control flow).
            accLo = __builtin_amdgcn_wmma_f32_16x16x32_f16(
                        false, A, false, __builtin_bit_cast(v16h, bLo),
                        (short)0, accLo, false, false);
            accHi = __builtin_amdgcn_wmma_f32_16x16x32_f16(
                        false, A, false, __builtin_bit_cast(v16h, bHi),
                        (short)0, accHi, false, false);
        }
    }

    // Scalar tail (empty when n % stride == 0): LDS-atomic histogram.
    const long long gsz = (long long)gridDim.x * blockDim.x;
    for (long long j = nMain + (long long)blockIdx.x * blockDim.x + tid; j < n; j += gsz) {
        float bce; int idx;
        ghm_elem(in[j], tg[j], bce, idx);
        atomicAdd(&sSum[idx], bce);      // ds_add_f32
        atomicAdd(&sCnt[idx], 1.0f);
    }

    // Fold WMMA accumulators into the block-shared bins.
    // Lanes 0-15 hold rows 0 (VGPR0) and 1 (VGPR1); lanes 16-31 hold rows 8,9 (zero).
    if (lane < 16) {
        atomicAdd(&sSum[n0],      accLo[0]);
        atomicAdd(&sCnt[n0],      accLo[1]);
        atomicAdd(&sSum[n0 + 16], accHi[0]);
        atomicAdd(&sCnt[n0 + 16], accHi[1]);
    }
    __syncthreads();
    if (tid < 32) {
        atomicAdd(&gSum[tid], sSum[tid]);   // global_atomic_add_f32
        atomicAdd(&gCnt[tid], sCnt[tid]);
    }
}

// Finalize: 30-bin momentum update + weighted-loss ratio -> scalar.
__global__ void ghm_finalize(const float* __restrict__ gSum,
                             const float* __restrict__ gCnt,
                             const float* __restrict__ acc_sum,
                             float* __restrict__ out, float tot) {
    int t = threadIdx.x;                 // 32 threads, one wave
    float numer = 0.0f, denom = 0.0f;
    if (t < GHM_BINS) {
        float cnt = gCnt[t];
        float sum = gSum[t];
        float acc = acc_sum[t];
        if (cnt > 0.0f) {
            float na = GHM_MOM * acc + (1.0f - GHM_MOM) * cnt;
            float bw = tot / na;
            numer = bw * sum;
            denom = bw * cnt;
        }
    }
    #pragma unroll
    for (int off = 16; off > 0; off >>= 1) {
        numer += __shfl_down(numer, off);
        denom += __shfl_down(denom, off);
    }
    // loss = sum(w*bce)/tot with w normalized by mean -> Σ bw·S / Σ bw·cnt
    if (t == 0) out[0] = numer / denom;
}

extern "C" void kernel_launch(void* const* d_in, const int* in_sizes, int n_in,
                              void* d_out, int out_size, void* d_ws, size_t ws_size,
                              hipStream_t stream) {
    const float* in  = (const float*)d_in[0];
    const float* tg  = (const float*)d_in[1];
    const float* acc = (const float*)d_in[2];
    float* out  = (float*)d_out;
    float* gSum = (float*)d_ws;          // 32 floats
    float* gCnt = gSum + 32;             // 32 floats
    long long n = (long long)in_sizes[0];

    ghm_init<<<1, 64, 0, stream>>>(gSum);
    ghm_pass1<<<2048, 256, 0, stream>>>(in, tg, gSum, gCnt, n);
    ghm_finalize<<<1, 32, 0, stream>>>(gSum, gCnt, acc, out, (float)n);
}
